// PaiNN_1382979469618
// MI455X (gfx1250) — matrix operand order repaired and checked by reading, not compiled, not run on hardware
//
#include <hip/hip_runtime.h>
#include <hip/hip_bf16.h>

// ---- problem constants (match reference) ----
#define N_ATOMS 10000
#define N_EDGES 100000
#define FDIM    128
#define N_INT   3
#define NUM_RBF 50
#define MAX_Z   100
#define KPAD    64               // RBF K padded 50 -> 64 for WMMA
#define FILT_N  (N_INT*3*FDIM)   // 1152
#define PI_F    3.14159265358979f

typedef __attribute__((ext_vector_type(16))) _Float16 v16h;
typedef __attribute__((ext_vector_type(8)))  _Float16 h8;
typedef __attribute__((ext_vector_type(8)))  float    v8f;

// ======================= WMMA fragment loaders ==========================
// ISA 7.12.2: 16-bit A 16x32 (MxK): lane m = lane&15; hi = (lane>>4)*8.
//   elements 0..7  <-> K = k0 + hi + e        (8 contiguous halfs)
//   elements 8..15 <-> K = k0 + 16 + hi + e   (8 contiguous halfs)
__device__ __forceinline__ v16h load_A_f16(const _Float16* A, int lda,
                                           int m0, int k0, int lane) {
  int m  = m0 + (lane & 15);
  int hi = (lane >> 4) * 8;
  const _Float16* row = A + (size_t)m * lda + k0 + hi;
  h8 lo = *(const h8*)(row);       // 16B aligned: k0%32==0, hi in {0,8}
  h8 hh = *(const h8*)(row + 16);
  v16h a;
#pragma unroll
  for (int e = 0; e < 8; ++e) { a[e] = lo[e]; a[8 + e] = hh[e]; }
  return a;
}

// B stored TRANSPOSED as Bt[N][K] (row-major, ldb = K).
// 16-bit B 32x16 (KxN): lane n = lane&15; kb = k0 + (lane>>4)*8;
// element j<8 <-> K = kb + j, j>=8 <-> K = kb + 16 + (j-8) (matches A).
__device__ __forceinline__ v16h load_Bt_f16(const _Float16* Bt, int ldb,
                                            int k0, int n0, int lane) {
  int n  = n0 + (lane & 15);
  int kb = k0 + (lane >> 4) * 8;
  const _Float16* row = Bt + (size_t)n * ldb + kb;
  h8 lo = *(const h8*)(row);
  h8 hh = *(const h8*)(row + 16);
  v16h b;
#pragma unroll
  for (int j = 0; j < 8; ++j) { b[j] = lo[j]; b[8 + j] = hh[j]; }
  return b;
}

__device__ __forceinline__ float silu_f(float v) {
  return v / (1.f + __expf(-v));
}

// ======================= generic WMMA GEMM ==============================
// C[M,N] = op( A[M,K](f16,row-major) * Bt[N,K]^T (f16) + bias[N] ) * rowscale[M]
// Compile-time K (fully unrolled -> K/32 back-to-back v_wmma), compile-time
// fuse flags (straight-line epilogue). M%16==0, N%16==0. One tile per wave.
template <int K, bool SILU, bool BIAS, bool RSCALE>
__global__ void k_gemm_wmma(const _Float16* __restrict__ A, int lda,
                            const _Float16* __restrict__ Bt,
                            float* __restrict__ C, int ldc,
                            const float* __restrict__ bias,
                            const float* __restrict__ rowscale,
                            int M, int N) {
  int lane  = threadIdx.x & 31;
  int wave  = threadIdx.x >> 5;
  int ntn   = N >> 4;
  int tile  = blockIdx.x * (blockDim.x >> 5) + wave;
  int total = (M >> 4) * ntn;
  if (tile >= total) return;          // wave-uniform: EXEC stays all-1s
  int mt = tile / ntn, nt = tile - mt * ntn;
  int m0 = mt << 4,  n0 = nt << 4;

  v8f acc = {};
#pragma unroll
  for (int k0 = 0; k0 < K; k0 += 32) {
    v16h a = load_A_f16(A, lda, m0, k0, lane);
    v16h b = load_Bt_f16(Bt, K, k0, n0, lane);
    acc = __builtin_amdgcn_wmma_f32_16x16x32_f16(
        false, a, false, b, (short)0, acc, false, false);
  }

  // C/D layout: lane n = n0+(lane&15); vgpr r -> row m0 + (lane>=16?8:0) + r
  int n     = n0 + (lane & 15);
  int mbase = m0 + (lane >> 4) * 8;
  float bv  = BIAS ? bias[n] : 0.f;
#pragma unroll
  for (int r = 0; r < 8; ++r) {
    int m   = mbase + r;
    float v = acc[r] + bv;
    if (SILU)   v = silu_f(v);
    if (RSCALE) v *= rowscale[m];
    C[(size_t)m * ldc + n] = v;
  }
}

// ======================= small helper kernels ==========================
__global__ void k_cvt_f16(const float* __restrict__ src,
                          _Float16* __restrict__ dst, int count) {
  int i = blockIdx.x * blockDim.x + threadIdx.x;
  if (i < count) dst[i] = (_Float16)src[i];
}

__global__ void k_zero_f32(float* __restrict__ dst, int count) {
  int i = blockIdx.x * blockDim.x + threadIdx.x;
  if (i < count) dst[i] = 0.f;
}

// Convert + transpose one weight matrix: W f32 [Kact,N] -> Wt f16 [N,Kpad],
// zero-padding rows Kact..Kpad-1.
__global__ void k_cvtT_f16(const float* __restrict__ W,
                           _Float16* __restrict__ Wt,
                           int Kact, int Kpad, int N) {
  int i = blockIdx.x * blockDim.x + threadIdx.x;
  if (i >= N * Kpad) return;
  int n = i / Kpad, k = i - n * Kpad;
  Wt[i] = (k < Kact) ? (_Float16)W[(size_t)k * N + n] : (_Float16)0.f;
}

// q = onehot @ emb_W + emb_b   (one block per atom, 128 threads)
__global__ void k_embed(const float* __restrict__ onehot,
                        const float* __restrict__ W,
                        const float* __restrict__ b,
                        float* __restrict__ q) {
  int a = blockIdx.x, f = threadIdx.x;
  const float* row = onehot + (size_t)a * MAX_Z;
  float s = b[f];
#pragma unroll 4
  for (int z = 0; z < MAX_Z; ++z) s += row[z] * W[(size_t)z * FDIM + f];
  q[(size_t)a * FDIM + f] = s;
}

// per-edge geometry: dir, fcut, RBF phi (f16, K padded to 64)
__global__ void k_edge_geom(const int* __restrict__ ii, const int* __restrict__ jj,
                            const float* __restrict__ pos,
                            _Float16* __restrict__ phi16,
                            float* __restrict__ fcut,
                            float* __restrict__ dir, int E) {
  int e = blockIdx.x * blockDim.x + threadIdx.x;
  if (e >= E) return;
  int i = ii[e], j = jj[e];
  float rx = pos[j*3+0] - pos[i*3+0];
  float ry = pos[j*3+1] - pos[i*3+1];
  float rz = pos[j*3+2] - pos[i*3+2];
  float d   = sqrtf(rx*rx + ry*ry + rz*rz);
  float inv = 1.f / d;
  dir[e*3+0] = rx*inv; dir[e*3+1] = ry*inv; dir[e*3+2] = rz*inv;
  fcut[e] = (d < 5.f) ? 0.5f * (__cosf(d * PI_F / 5.f) + 1.f) : 0.f;
  const float step  = 5.f / 49.f;
  const float coeff = -0.5f / (step * step);
  _Float16* p = phi16 + (size_t)e * KPAD;
#pragma unroll 8
  for (int k = 0; k < KPAD; ++k) {
    if (k < NUM_RBF) {
      float t = d - step * (float)k;
      p[k] = (_Float16)__expf(coeff * t * t);
    } else {
      p[k] = (_Float16)0.f;
    }
  }
}

// edge message: xj = x[idx_j]*filt; scatter-add dq to q[idx_i],
// dmu = dmuR*dir + dmumu*mu_old[idx_j] to mu[idx_i]. One block per edge.
__global__ void k_edge_msg(const int* __restrict__ ii, const int* __restrict__ jj,
                           const float* __restrict__ x,
                           const float* __restrict__ filt,
                           const float* __restrict__ dir,
                           const float* __restrict__ mu_old,
                           float* __restrict__ q, float* __restrict__ mu) {
  int e = blockIdx.x, f = threadIdx.x;
  int i = ii[e], j = jj[e];
  const float* xr = x    + (size_t)j * (3*FDIM);
  const float* fr = filt + (size_t)e * (3*FDIM);
  float dq = xr[f]          * fr[f];
  float dR = xr[FDIM + f]   * fr[FDIM + f];
  float dm = xr[2*FDIM + f] * fr[2*FDIM + f];
  atomicAdd(&q[(size_t)i * FDIM + f], dq);
#pragma unroll
  for (int c = 0; c < 3; ++c) {
    float v = dR * dir[e*3 + c] + dm * mu_old[((size_t)j*3 + c) * FDIM + f];
    atomicAdd(&mu[((size_t)i*3 + c) * FDIM + f], v);
  }
}

// build ctx = [q, ||mu_V||] from mumix [n*3, 256] (V = cols 0..127)
__global__ void k_norm_ctx(const float* __restrict__ q,
                           const float* __restrict__ mumix,
                           float* __restrict__ ctx) {
  int a = blockIdx.x, f = threadIdx.x;
  float s = 1e-8f;
#pragma unroll
  for (int c = 0; c < 3; ++c) {
    float v = mumix[((size_t)(a*3 + c)) * (2*FDIM) + f];
    s += v * v;
  }
  ctx[(size_t)a * (2*FDIM) + f]        = q[(size_t)a * FDIM + f];
  ctx[(size_t)a * (2*FDIM) + FDIM + f] = sqrtf(s);
}

// final mixing update: q += dq + dqmu * sum_c(muV*muW); mu += dmu_i * muW
__global__ void k_mix_update(const float* __restrict__ mumix,
                             const float* __restrict__ y,
                             float* __restrict__ q, float* __restrict__ mu) {
  int a = blockIdx.x, f = threadIdx.x;
  const float* yr = y + (size_t)a * (3*FDIM);
  float dmu_i = yr[FDIM + f];
  float s = 0.f;
#pragma unroll
  for (int c = 0; c < 3; ++c) {
    const float* row = mumix + ((size_t)(a*3 + c)) * (2*FDIM);
    float v = row[f], w = row[FDIM + f];
    s += v * w;
    mu[((size_t)a*3 + c) * FDIM + f] += dmu_i * w;
  }
  q[(size_t)a * FDIM + f] += yr[f] + yr[2*FDIM + f] * s;
}

// ======================= host side ==============================
static inline size_t align256(size_t x) { return (x + 255) & ~(size_t)255; }

template <int K, bool SILU, bool BIAS, bool RSCALE>
static void launch_gemm(const _Float16* A, int lda, const _Float16* Bt,
                        float* C, int ldc, const float* bias,
                        const float* rowscale, int M, int N,
                        hipStream_t stream) {
  int tiles  = (M >> 4) * (N >> 4);
  int waves  = 8;                       // 256 threads per block
  int blocks = (tiles + waves - 1) / waves;
  k_gemm_wmma<K, SILU, BIAS, RSCALE>
      <<<blocks, waves * 32, 0, stream>>>(A, lda, Bt, C, ldc, bias, rowscale,
                                          M, N);
}

extern "C" void kernel_launch(void* const* d_in, const int* in_sizes, int n_in,
                              void* d_out, int out_size, void* d_ws, size_t ws_size,
                              hipStream_t stream) {
  (void)in_sizes; (void)n_in; (void)out_size; (void)ws_size;
  const float* onehot = (const float*)d_in[0];
  const float* pos    = (const float*)d_in[1];
  const int*   eidx   = (const int*)  d_in[2];
  const float* emb_W  = (const float*)d_in[3];
  const float* emb_b  = (const float*)d_in[4];
  const float* filt_W = (const float*)d_in[5];
  const float* filt_b = (const float*)d_in[6];
  const float* int_W1 = (const float*)d_in[7];
  const float* int_b1 = (const float*)d_in[8];
  const float* int_W2 = (const float*)d_in[9];
  const float* int_b2 = (const float*)d_in[10];
  const float* mix_Wmu= (const float*)d_in[11];
  const float* mix_W1 = (const float*)d_in[12];
  const float* mix_b1 = (const float*)d_in[13];
  const float* mix_W2 = (const float*)d_in[14];
  const float* mix_b2 = (const float*)d_in[15];

  const int* idx_i = eidx;            // [E]
  const int* idx_j = eidx + N_EDGES;  // [E]

  // state lives in d_out: q [n,128] then mu [n,3,128]
  float* q  = (float*)d_out;
  float* mu = q + (size_t)N_ATOMS * FDIM;

  // ---- workspace carve-up ----
  char* ws = (char*)d_ws;
  size_t off = 0;
  _Float16* phi16   = (_Float16*)(ws + off); off = align256(off + (size_t)N_EDGES * KPAD * 2);
  float*    fcut    = (float*)   (ws + off); off = align256(off + (size_t)N_EDGES * 4);
  float*    dirb    = (float*)   (ws + off); off = align256(off + (size_t)N_EDGES * 3 * 4);
  _Float16* filtWT  = (_Float16*)(ws + off); off = align256(off + (size_t)FILT_N * KPAD * 2);
  _Float16* W16     = (_Float16*)(ws + off); off = align256(off + (size_t)540672 * 2);
  float*    filt    = (float*)   (ws + off); off = align256(off + (size_t)N_EDGES * 3*FDIM * 4);
  float*    xbuf    = (float*)   (ws + off); off = align256(off + (size_t)N_ATOMS * 3*FDIM * 4);
  float*    hbuf    = (float*)   (ws + off); off = align256(off + (size_t)N_ATOMS * FDIM * 4);
  float*    mumix   = (float*)   (ws + off); off = align256(off + (size_t)N_ATOMS * 3 * 2*FDIM * 4);
  float*    ctx     = (float*)   (ws + off); off = align256(off + (size_t)N_ATOMS * 2*FDIM * 4);
  float*    ybuf    = (float*)   (ws + off); off = align256(off + (size_t)N_ATOMS * 3*FDIM * 4);
  _Float16* act16   = (_Float16*)(ws + off); off = align256(off + (size_t)30000 * 256 * 2);
  float*    mu_old  = (float*)   (ws + off); off = align256(off + (size_t)N_ATOMS * 3*FDIM * 4);

  // transposed f16 weight slabs inside W16 (element offsets); each layer
  // stores N_INT matrices of [N, K] halfs.
  _Float16* W1_T  = W16;                    // 3 x [128,128]
  _Float16* W2_T  = W1_T  + 3 * 128 * 128;  // 3 x [384,128]
  _Float16* WmuT  = W2_T  + 3 * 384 * 128;  // 3 x [256,128]
  _Float16* mW1_T = WmuT  + 3 * 256 * 128;  // 3 x [128,256]
  _Float16* mW2_T = mW1_T + 3 * 128 * 256;  // 3 x [384,128]

  // ---- one-time setup ----
  k_embed<<<N_ATOMS, FDIM, 0, stream>>>(onehot, emb_W, emb_b, q);
  {
    int cnt = N_ATOMS * 3 * FDIM;
    k_zero_f32<<<(cnt + 255) / 256, 256, 0, stream>>>(mu, cnt);
  }
  k_edge_geom<<<(N_EDGES + 255) / 256, 256, 0, stream>>>(idx_i, idx_j, pos,
                                                         phi16, fcut, dirb, N_EDGES);
  // filt_W [50,1152] -> filtWT [1152,64] (zero-padded K)
  {
    int cnt = FILT_N * KPAD;
    k_cvtT_f16<<<(cnt + 255) / 256, 256, 0, stream>>>(filt_W, filtWT,
                                                      NUM_RBF, KPAD, FILT_N);
  }
  // per-interaction weight transposes (W[K,N] -> Wt[N,K], f16)
  for (int it = 0; it < N_INT; ++it) {
    int c1 = 128 * 128, c2 = 384 * 128, c3 = 256 * 128, c4 = 128 * 256;
    k_cvtT_f16<<<(c1 + 255) / 256, 256, 0, stream>>>(
        int_W1 + (size_t)it * 128 * 128, W1_T + (size_t)it * c1, 128, 128, 128);
    k_cvtT_f16<<<(c2 + 255) / 256, 256, 0, stream>>>(
        int_W2 + (size_t)it * 128 * 384, W2_T + (size_t)it * c2, 128, 128, 384);
    k_cvtT_f16<<<(c3 + 255) / 256, 256, 0, stream>>>(
        mix_Wmu + (size_t)it * 128 * 256, WmuT + (size_t)it * c3, 128, 128, 256);
    k_cvtT_f16<<<(c4 + 255) / 256, 256, 0, stream>>>(
        mix_W1 + (size_t)it * 256 * 128, mW1_T + (size_t)it * c4, 256, 256, 128);
    k_cvtT_f16<<<(c2 + 255) / 256, 256, 0, stream>>>(
        mix_W2 + (size_t)it * 128 * 384, mW2_T + (size_t)it * c2, 128, 128, 384);
  }

  // ---- interaction loop ----
  for (int it = 0; it < N_INT; ++it) {
    // h = silu(q @ W1 + b1)
    k_cvt_f16<<<(N_ATOMS*FDIM + 255)/256, 256, 0, stream>>>(q, act16, N_ATOMS*FDIM);
    launch_gemm<128, true, true, false>(act16, FDIM,
        W1_T + (size_t)it * 128 * 128, hbuf, FDIM,
        int_b1 + it * FDIM, nullptr, N_ATOMS, FDIM, stream);
    // x = h @ W2 + b2
    k_cvt_f16<<<(N_ATOMS*FDIM + 255)/256, 256, 0, stream>>>(hbuf, act16, N_ATOMS*FDIM);
    launch_gemm<128, false, true, false>(act16, FDIM,
        W2_T + (size_t)it * 384 * 128, xbuf, 3*FDIM,
        int_b2 + it * 3*FDIM, nullptr, N_ATOMS, 3*FDIM, stream);
    // filters_i = (phi @ filt_W[:, it*384:(it+1)*384] + b) * fcut
    launch_gemm<KPAD, false, true, true>(phi16, KPAD,
        filtWT + (size_t)(it * 3*FDIM) * KPAD, filt, 3*FDIM,
        filt_b + it * 3*FDIM, fcut, N_EDGES, 3*FDIM, stream);
    // snapshot mu, then edge scatter
    hipMemcpyAsync(mu_old, mu, (size_t)N_ATOMS*3*FDIM*sizeof(float),
                   hipMemcpyDeviceToDevice, stream);
    k_edge_msg<<<N_EDGES, FDIM, 0, stream>>>(idx_i, idx_j, xbuf, filt, dirb,
                                             mu_old, q, mu);
    // mumix = mu @ Wmu   (M = 3n rows)
    k_cvt_f16<<<(N_ATOMS*3*FDIM + 255)/256, 256, 0, stream>>>(mu, act16, N_ATOMS*3*FDIM);
    launch_gemm<128, false, false, false>(act16, FDIM,
        WmuT + (size_t)it * 256 * 128, mumix, 2*FDIM,
        nullptr, nullptr, 3*N_ATOMS, 2*FDIM, stream);
    // ctx = [q, ||mu_V||]
    k_norm_ctx<<<N_ATOMS, FDIM, 0, stream>>>(q, mumix, ctx);
    // h2 = silu(ctx @ mix_W1 + b1)
    k_cvt_f16<<<(N_ATOMS*2*FDIM + 255)/256, 256, 0, stream>>>(ctx, act16, N_ATOMS*2*FDIM);
    launch_gemm<256, true, true, false>(act16, 2*FDIM,
        mW1_T + (size_t)it * 128 * 256, hbuf, FDIM,
        mix_b1 + it * FDIM, nullptr, N_ATOMS, FDIM, stream);
    // y = h2 @ mix_W2 + b2
    k_cvt_f16<<<(N_ATOMS*FDIM + 255)/256, 256, 0, stream>>>(hbuf, act16, N_ATOMS*FDIM);
    launch_gemm<128, false, true, false>(act16, FDIM,
        mW2_T + (size_t)it * 384 * 128, ybuf, 3*FDIM,
        mix_b2 + it * 3*FDIM, nullptr, N_ATOMS, 3*FDIM, stream);
    // final state update
    k_mix_update<<<N_ATOMS, FDIM, 0, stream>>>(mumix, ybuf, q, mu);
  }
}
